// ROEN_Final_73443940761984
// MI455X (gfx1250) — compile-verified, matrix operand order checked by reading, not run on hardware
//
#include <hip/hip_runtime.h>
#include <hip/hip_bf16.h>

// ---------------------------------------------------------------------------
// MI455X (gfx1250) implementation.
//  - All dense GEMMs -> v_wmma_f32_16x16x32_f16 (f16 in, f32 accumulate).
//  - Wave tile 32x64: 2 A-frags x 4 B-frags -> 8 WMMAs per k-step (fragment
//    reuse in both dims, batched dscnt waits).
//  - A stream (HBM) software-pipelined through VGPRs (6 of 8 float4s; tail 2
//    loaded at stage time from already-hot cache lines -> -8 VGPRs, no spills).
//  - W staged directly (L2-resident: 192MB L2 holds every weight matrix).
//  - Edge concat [h[src], h[dst], e] fused into the GEMM A-tile gather.
//  - Segment softmax / aggregation: coalesced wave-per-edge + L2 atomics.
//  - Temporal linear attention: one wave32 per (node, head), KV outer product
//    kept in registers via cross-lane shuffles.
// ---------------------------------------------------------------------------

typedef __attribute__((ext_vector_type(16))) _Float16 v16h;
typedef __attribute__((ext_vector_type(8)))  float    v8f;

#define BM  128   // M-tile per block
#define KCH 64    // K chunk staged in LDS
#define KCP 72    // padded LDS row stride (halves): 8-half aligned, no bank clash
#define PFV 6     // float4s of the A strip prefetched into VGPRs (of 8)

__device__ __forceinline__ float wave_sum32(float v) {
  #pragma unroll
  for (int o = 16; o > 0; o >>= 1) v += __shfl_xor(v, o, 32);
  return v;
}

// ---------------------------------------------------------------------------
// Generic WMMA GEMM:  C[M, Nout] = act( A[M, K] * W[Nout, K]^T + bias ) + post
// segW == 0  : single source A0 (row stride K)
// segW == 128: A row r = concat( A0[I0 ? I0[r] : r], A1[...], A2[...] ), K=384
// act: 0 none, 1 elu(x)+1, 2 relu
// ---------------------------------------------------------------------------
__global__ __launch_bounds__(256, 1) void gemm_wmma_kernel(
    const float* __restrict__ A0, const int* __restrict__ I0,
    const float* __restrict__ A1, const int* __restrict__ I1,
    const float* __restrict__ A2, const int* __restrict__ I2,
    int segW,
    const float* __restrict__ W, const float* __restrict__ bias,
    const float* __restrict__ post,
    float* __restrict__ C,
    int M, int K, int Nout, int act)
{
  __shared__ _Float16 As[BM * KCP];
  __shared__ _Float16 Ws[BM * KCP];

  const int tid   = threadIdx.x;
  const int lane  = tid & 31;
  const int wave  = tid >> 5;
  const int rg    = wave >> 1;         // row group: 32 rows
  const int cg    = wave & 1;          // col group: 64 cols
  const int mBase = blockIdx.x * BM;
  const int nBase = blockIdx.y * BM;
  const int r     = tid >> 1;          // staging row handled by this thread
  const int c0    = (tid & 1) * 32;    // staging column half

  v8f acc[2][4];
  #pragma unroll
  for (int m = 0; m < 2; m++)
    #pragma unroll
    for (int n = 0; n < 4; n++) acc[m][n] = v8f{0,0,0,0,0,0,0,0};

  // ---- loop-invariant addressing (hoisted: no per-chunk index loads) ----
  const int grow = mBase + r;
  const float* rp0 = nullptr;
  const float* rp1 = nullptr;
  const float* rp2 = nullptr;
  if (grow < M) {
    if (segW) {                        // segW == 128: concat-gather row pointers
      rp0 = A0 + (size_t)(I0 ? I0[grow] : grow) * 128;
      rp1 = A1 + (size_t)(I1 ? I1[grow] : grow) * 128;
      rp2 = A2 + (size_t)(I2 ? I2[grow] : grow) * 128;
    } else {
      rp0 = A0 + (size_t)grow * K;
    }
  }
  const float* wrow = (nBase + r < Nout) ? (W + (size_t)(nBase + r) * K) : nullptr;
  _Float16* const da = &As[r * KCP + c0];
  _Float16* const dw = &Ws[r * KCP + c0];

  // current-chunk A strip pointer (per thread): srcp + cbase
  auto rowA = [&](int kc, const float*& srcp, int& cbase) {
    if (segW) {                        // chunk lies entirely within one 128-wide segment
      const int seg = kc >> 7;
      srcp  = (seg == 0) ? rp0 : ((seg == 1) ? rp1 : rp2);
      cbase = (kc & 127) + c0;
    } else {
      srcp  = (kc + c0 < K) ? rp0 : nullptr;   // K multiple of 32; c0 steps by 32
      cbase = kc + c0;
    }
  };

  float4 ra[PFV];   // prefetched A strip head (hides HBM latency)

  auto fetchA = [&](int kc) {
    const float* srcp; int cbase;
    rowA(kc, srcp, cbase);
    if (srcp) {
      const float4* s4 = reinterpret_cast<const float4*>(srcp + cbase);
      #pragma unroll
      for (int ii = 0; ii < PFV; ii++) ra[ii] = s4[ii];
    } else {
      #pragma unroll
      for (int ii = 0; ii < PFV; ii++) ra[ii] = make_float4(0.f, 0.f, 0.f, 0.f);
    }
  };

  auto stage = [&](int kc) {   // park A prefetch (+ hot-line tail) + stream W into LDS
    const float* srcp; int cbase;
    rowA(kc, srcp, cbase);
    float4 rt[8 - PFV];        // tail: same cache lines as prefetched head -> L0 hit
    if (srcp) {
      const float4* s4 = reinterpret_cast<const float4*>(srcp + cbase);
      #pragma unroll
      for (int ii = PFV; ii < 8; ii++) rt[ii - PFV] = s4[ii];
    } else {
      #pragma unroll
      for (int ii = PFV; ii < 8; ii++) rt[ii - PFV] = make_float4(0.f, 0.f, 0.f, 0.f);
    }
    #pragma unroll
    for (int ii = 0; ii < PFV; ii++) {
      da[4*ii+0] = (_Float16)ra[ii].x; da[4*ii+1] = (_Float16)ra[ii].y;
      da[4*ii+2] = (_Float16)ra[ii].z; da[4*ii+3] = (_Float16)ra[ii].w;
    }
    #pragma unroll
    for (int ii = PFV; ii < 8; ii++) {
      const float4 v = rt[ii - PFV];
      da[4*ii+0] = (_Float16)v.x; da[4*ii+1] = (_Float16)v.y;
      da[4*ii+2] = (_Float16)v.z; da[4*ii+3] = (_Float16)v.w;
    }
    if (wrow && kc + c0 < K) {
      const float4* s4 = reinterpret_cast<const float4*>(wrow + kc + c0);
      #pragma unroll
      for (int ii = 0; ii < 8; ii++) {
        float4 v = s4[ii];
        dw[4*ii+0] = (_Float16)v.x; dw[4*ii+1] = (_Float16)v.y;
        dw[4*ii+2] = (_Float16)v.z; dw[4*ii+3] = (_Float16)v.w;
      }
    } else {
      #pragma unroll
      for (int ii = 0; ii < 32; ii++) dw[ii] = (_Float16)0.f;
    }
  };

  fetchA(0);                             // prologue prefetch

  for (int kc = 0; kc < K; kc += KCH) {
    stage(kc);
    __syncthreads();                     // tile visible to all waves

    const int nxt = kc + KCH;
    if (nxt < K) fetchA(nxt);            // HBM latency hidden under WMMAs

    const int koff = (lane >> 4) * 8;    // ISA 16-bit A/B lane layout
    const int cl   = lane & 15;
    #pragma unroll
    for (int ks = 0; ks < KCH; ks += 32) {
      v16h af[2];
      #pragma unroll
      for (int m = 0; m < 2; m++) {
        const _Float16* ap = &As[(rg * 32 + m * 16 + cl) * KCP + ks + koff];
        #pragma unroll
        for (int i = 0; i < 8; i++) { af[m][i] = ap[i]; af[m][i + 8] = ap[i + 16]; }
      }
      v16h bf[4];                        // batch all B loads before the WMMA chain
      #pragma unroll
      for (int n = 0; n < 4; n++) {
        const _Float16* bp = &Ws[(cg * 64 + n * 16 + cl) * KCP + ks + koff];
        #pragma unroll
        for (int i = 0; i < 8; i++) { bf[n][i] = bp[i]; bf[n][i + 8] = bp[i + 16]; }
      }
      #pragma unroll
      for (int m = 0; m < 2; m++)
        #pragma unroll
        for (int n = 0; n < 4; n++)
          acc[m][n] = __builtin_amdgcn_wmma_f32_16x16x32_f16(
              false, af[m], false, bf[n], (short)0, acc[m][n], false, false);
    }
    __syncthreads();                     // LDS reuse guard before next stage()
  }

  // ---- epilogue: bias + activation + optional residual ----
  const int cl = lane & 15;
  #pragma unroll
  for (int m = 0; m < 2; m++) {
    const int rbase = mBase + rg * 32 + m * 16 + ((lane >> 4) << 3);
    #pragma unroll
    for (int n = 0; n < 4; n++) {
      const int col = nBase + cg * 64 + n * 16 + cl;
      if (col >= Nout) continue;
      const float bv = bias ? bias[col] : 0.f;
      #pragma unroll
      for (int i = 0; i < 8; i++) {
        const int row = rbase + i;
        if (row < M) {
          float y = acc[m][n][i] + bv;
          if (act == 1)      y = (y > 0.f) ? (y + 1.f) : __expf(y);  // elu(x)+1
          else if (act == 2) y = fmaxf(y, 0.f);
          const size_t o = (size_t)row * Nout + col;
          if (post) y += post[o];
          C[o] = y;
        }
      }
    }
  }
}

// ---------------------------------------------------------------------------
// Row LayerNorm (+opt pre-residual, opt indexed pre-residual rows, activation,
// opt post-residual).  blockDim.x == D (128 or 256), one row per block.
// act: 0 none, 1 exact gelu, 2 relu
// ---------------------------------------------------------------------------
__global__ void ln_act_kernel(
    const float* __restrict__ in, const float* __restrict__ pre,
    const int* __restrict__ pre_idx,
    const float* __restrict__ g, const float* __restrict__ b,
    const float* __restrict__ post, float* __restrict__ out,
    int D, int act)
{
  __shared__ float red[256];
  const int r = blockIdx.x, c = threadIdx.x;
  const size_t base = (size_t)r * D;

  float v = in[base + c];
  if (pre) {
    const int pr = pre_idx ? pre_idx[r] : r;
    v += pre[(size_t)pr * D + c];
  }
  red[c] = v; __syncthreads();
  for (int s = D >> 1; s > 0; s >>= 1) { if (c < s) red[c] += red[c + s]; __syncthreads(); }
  const float mean = red[0] / D; __syncthreads();
  const float d0 = v - mean;
  red[c] = d0 * d0; __syncthreads();
  for (int s = D >> 1; s > 0; s >>= 1) { if (c < s) red[c] += red[c + s]; __syncthreads(); }
  const float var = red[0] / D;

  float y = d0 * rsqrtf(var + 1e-5f) * g[c] + b[c];
  if (act == 1)      y = 0.5f * y * (1.f + erff(y * 0.70710678118654752f));
  else if (act == 2) y = fmaxf(y, 0.f);
  if (post) y += post[base + c];
  out[base + c] = y;
}

// ---------------------------------------------------------------------------
// Edge attention: per-edge score + order-preserving atomic max per (dst, head)
// ---------------------------------------------------------------------------
__global__ __launch_bounds__(256) void edge_score_kernel(
    const float* __restrict__ q, const float* __restrict__ k,
    const float* __restrict__ ee,
    const int* __restrict__ src, const int* __restrict__ dst,
    float* __restrict__ score, unsigned* __restrict__ menc, int E)
{
  const int wave = threadIdx.x >> 5, lane = threadIdx.x & 31;
  const int e = blockIdx.x * 8 + wave;
  if (e >= E) return;
  const int s = src[e], d = dst[e];
  const size_t qb = (size_t)d * 128, kb = (size_t)s * 128, eb = (size_t)e * 128;
  float p0 = q[qb + lane]      * (k[kb + lane]      + ee[eb + lane]);
  float p1 = q[qb + lane + 32] * (k[kb + lane + 32] + ee[eb + lane + 32]);
  float p2 = q[qb + lane + 64] * (k[kb + lane + 64] + ee[eb + lane + 64]);
  float p3 = q[qb + lane + 96] * (k[kb + lane + 96] + ee[eb + lane + 96]);
  p0 = wave_sum32(p0); p1 = wave_sum32(p1);
  p2 = wave_sum32(p2); p3 = wave_sum32(p3);
  if (lane == 0) {
    const float sc = 0.17677669529663687f;  // 1/sqrt(32)
    float v[4] = {p0 * sc, p1 * sc, p2 * sc, p3 * sc};
    #pragma unroll
    for (int j = 0; j < 4; j++) {
      score[(size_t)e * 4 + j] = v[j];
      unsigned u = __float_as_uint(v[j]);
      u = (u & 0x80000000u) ? ~u : (u | 0x80000000u);   // orderable encoding
      atomicMax(&menc[(size_t)d * 4 + j], u);
    }
  }
}

__global__ void edge_expsum_kernel(
    const int* __restrict__ dst, float* __restrict__ score,
    const unsigned* __restrict__ menc, float* __restrict__ den, int E)
{
  const int gid = blockIdx.x * blockDim.x + threadIdx.x;
  if (gid >= E * 4) return;
  const int e = gid >> 2, j = gid & 3;
  const int d = dst[e];
  const unsigned u = menc[d * 4 + j];
  const float m = (u & 0x80000000u) ? __uint_as_float(u ^ 0x80000000u)
                                    : __uint_as_float(~u);
  const float x = __expf(score[gid] - m);
  score[gid] = x;                       // reuse buffer for exp values
  atomicAdd(&den[d * 4 + j], x);
}

__global__ __launch_bounds__(256) void edge_aggregate_kernel(
    const float* __restrict__ v, const float* __restrict__ ee,
    const float* __restrict__ ex, const float* __restrict__ den,
    const int* __restrict__ src, const int* __restrict__ dst,
    float* __restrict__ agg, int E)
{
  const int wave = threadIdx.x >> 5, lane = threadIdx.x & 31;
  const int e = blockIdx.x * 8 + wave;
  if (e >= E) return;
  const int s = src[e], d = dst[e];
  float al = 0.f;
  if (lane < 4) al = ex[(size_t)e * 4 + lane] / (den[(size_t)d * 4 + lane] + 1e-16f);
  const float a0 = __shfl(al, 0, 32), a1 = __shfl(al, 1, 32);
  const float a2 = __shfl(al, 2, 32), a3 = __shfl(al, 3, 32);
  const size_t vb = (size_t)s * 128, eb = (size_t)e * 128, ab = (size_t)d * 128;
  atomicAdd(&agg[ab + lane],      a0 * (v[vb + lane]      + ee[eb + lane]));
  atomicAdd(&agg[ab + lane + 32], a1 * (v[vb + lane + 32] + ee[eb + lane + 32]));
  atomicAdd(&agg[ab + lane + 64], a2 * (v[vb + lane + 64] + ee[eb + lane + 64]));
  atomicAdd(&agg[ab + lane + 96], a3 * (v[vb + lane + 96] + ee[eb + lane + 96]));
}

// ---------------------------------------------------------------------------
// Subnet pooling
// ---------------------------------------------------------------------------
__global__ void subnet_accum_kernel(const float* __restrict__ h,
                                    const int* __restrict__ sid,
                                    float* __restrict__ sf,
                                    float* __restrict__ cnt, int N)
{
  const int gid = blockIdx.x * blockDim.x + threadIdx.x;
  if (gid >= N * 128) return;
  const int n = gid >> 7, c = gid & 127;
  const int s = sid[n];
  atomicAdd(&sf[s * 128 + c], h[(size_t)n * 128 + c]);
  if (c == 0) atomicAdd(&cnt[s], 1.0f);
}

__global__ void subnet_mean_kernel(float* __restrict__ sf,
                                   const float* __restrict__ cnt)
{
  const int gid = blockIdx.x * blockDim.x + threadIdx.x;  // 64*128 threads
  sf[gid] /= fmaxf(cnt[gid >> 7], 1.0f);
}

// ---------------------------------------------------------------------------
// Temporal: tpe add + depthwise conv (k=3, pad=1) + residual -> xp [T,N,128]
// ---------------------------------------------------------------------------
__global__ void conv_tpe_kernel(const float* __restrict__ nf,
                                const float* __restrict__ tpe,
                                const float* __restrict__ cw,
                                const float* __restrict__ cb,
                                float* __restrict__ xp, int N)
{
  const int gid = blockIdx.x * blockDim.x + threadIdx.x;
  if (gid >= N * 128) return;
  const int c = gid & 127, n = gid >> 7;
  float d[8];
  #pragma unroll
  for (int t = 0; t < 8; t++)
    d[t] = nf[(size_t)t * N * 128 + (size_t)n * 128 + c] + tpe[t * 128 + c];
  const float w0 = cw[c * 3], w1 = cw[c * 3 + 1], w2 = cw[c * 3 + 2], bb = cb[c];
  #pragma unroll
  for (int t = 0; t < 8; t++) {
    const float l = (t > 0) ? d[t - 1] : 0.f;
    const float r = (t < 7) ? d[t + 1] : 0.f;
    xp[(size_t)t * N * 128 + (size_t)n * 128 + c] = w0 * l + w1 * d[t] + w2 * r + bb + d[t];
  }
}

// ---------------------------------------------------------------------------
// Linear temporal attention: one wave32 per (node, head); KV (32x32) in regs.
// ---------------------------------------------------------------------------
__global__ __launch_bounds__(256) void temporal_attn_kernel(
    const float* __restrict__ tq, const float* __restrict__ tk,
    const float* __restrict__ tv, float* __restrict__ outp, int N)
{
  const int wave = threadIdx.x >> 5, lane = threadIdx.x & 31;
  const int g = blockIdx.x * 8 + wave;
  if (g >= N * 4) return;
  const int n = g >> 2, h = g & 3;
  const size_t base0 = (size_t)n * 128 + h * 32 + lane;

  float qt[8], kt[8], vt[8];
  #pragma unroll
  for (int t = 0; t < 8; t++) {
    const size_t b = (size_t)t * N * 128 + base0;
    qt[t] = tq[b]; kt[t] = tk[b]; vt[t] = tv[b];
  }
  float ks = 0.f;
  #pragma unroll
  for (int t = 0; t < 8; t++) ks += kt[t];

  float kv[32];
  #pragma unroll
  for (int d = 0; d < 32; d++) kv[d] = 0.f;
  #pragma unroll
  for (int t = 0; t < 8; t++) {
    #pragma unroll
    for (int d = 0; d < 32; d++) kv[d] += __shfl(kt[t], d, 32) * vt[t];
  }
  #pragma unroll
  for (int t = 0; t < 8; t++) {
    const float z = wave_sum32(qt[t] * ks);
    float num = 0.f;
    #pragma unroll
    for (int d = 0; d < 32; d++) num += __shfl(qt[t], d, 32) * kv[d];
    outp[(size_t)t * N * 128 + base0] = num / (z + 1e-6f);
  }
}

// ---------------------------------------------------------------------------
// Classifier head: preds[e, 0:2] = h1[e, 0:256] @ w2[2,256]^T + b2
// ---------------------------------------------------------------------------
__global__ __launch_bounds__(256) void cls_out_kernel(
    const float* __restrict__ h1, const float* __restrict__ w2,
    const float* __restrict__ b2, float* __restrict__ out, int E)
{
  const int wave = threadIdx.x >> 5, lane = threadIdx.x & 31;
  const int e = blockIdx.x * 8 + wave;
  if (e >= E) return;
  float a0 = 0.f, a1 = 0.f;
  #pragma unroll
  for (int j = 0; j < 8; j++) {
    const int c = lane + 32 * j;
    const float hv = h1[(size_t)e * 256 + c];
    a0 += hv * w2[c];
    a1 += hv * w2[256 + c];
  }
  a0 = wave_sum32(a0); a1 = wave_sum32(a1);
  if (lane == 0) {
    out[(size_t)e * 2]     = a0 + b2[0];
    out[(size_t)e * 2 + 1] = a1 + b2[1];
  }
}

// ---------------------------------------------------------------------------
// Host driver
// ---------------------------------------------------------------------------
extern "C" void kernel_launch(void* const* d_in, const int* in_sizes, int n_in,
                              void* d_out, int out_size, void* d_ws, size_t ws_size,
                              hipStream_t stream) {
  (void)in_sizes; (void)n_in; (void)out_size; (void)ws_size;
  const int T = 8, N = 20000, E = 160000;

  auto F = [&](int i) { return (const float*)d_in[i]; };
  // jax-pytree (sorted dict key) flattening of setup_inputs():
  const float* x       = F(0);      // [T,N,64]
  const float* ea      = F(1);      // [T,E,32]
  const float* cls_b1  = F(2);
  const float* cls_b2  = F(3);
  const float* cls_bn  = F(4);
  const float* cls_g   = F(5);
  const float* cls_w1  = F(6);      // [256,384]
  const float* cls_w2  = F(7);      // [2,256]
  const float* conv_b  = F(8);
  const float* conv_w  = F(9);      // [128,1,3]
  const float* eenc_b  = F(10);
  const float* eenc_w  = F(11);     // [128,32]
  // layers[l]: att{b,g,ob,ow,we,wk,wq,wv}, eu{b1,b2,bb1,bb2,g1,g2,w1,w2},
  //            sub{b,b1,b2,g,w1,w2}  -> base 12 + 22*l
  const float* ne_b    = F(56);
  const float* ne_w    = F(57);     // [128,64]
  const float* ta_kb   = F(58);
  const float* ta_kw   = F(59);
  const float* ta_ob   = F(60);
  const float* ta_ow   = F(61);
  const float* ta_qb   = F(62);
  const float* ta_qw   = F(63);
  const float* ta_vb   = F(64);
  const float* ta_vw   = F(65);
  const float* tpe     = F(66);     // [T,128]
  const int*   eidx    = (const int*)d_in[67];  // [T,2,E]
  const int*   snid    = (const int*)d_in[68];  // [T,N]

  // ---- workspace arena (floats) ----
  float* ws = (float*)d_ws;
  size_t off = 0;
  auto alloc = [&](size_t nelem) { float* p = ws + off; off += nelem; return p; };
  float* nf    = alloc((size_t)T * N * 128);   // node features per frame (h lives here)
  float* ef    = alloc((size_t)T * E * 128);   // edge features per frame
  float* qb_   = alloc((size_t)N * 128);
  float* kb_   = alloc((size_t)N * 128);
  float* vb_   = alloc((size_t)N * 128);
  float* eeb   = alloc((size_t)E * 128);       // ee proj, reused as eu z1
  float* score = alloc((size_t)E * 4);         // scores, then exp values
  unsigned* menc = (unsigned*)alloc((size_t)N * 4);
  float* den   = alloc((size_t)N * 4);
  float* agg   = alloc((size_t)N * 128);
  float* tmpN  = alloc((size_t)N * 128);
  float* tmpE  = alloc((size_t)E * 128);
  float* sf    = alloc(64 * 128);
  float* sft   = alloc(64 * 128);
  float* sf2   = alloc(64 * 128);
  float* cnt   = alloc(64);
  float* xp    = alloc((size_t)T * N * 128);
  float* tq    = alloc((size_t)T * N * 128);
  float* tk    = alloc((size_t)T * N * 128);
  float* tv    = alloc((size_t)T * N * 128);
  float* tat   = alloc((size_t)T * N * 128);
  float* h1    = alloc((size_t)E * 256);

  auto gemm = [&](const float* A, const float* W, const float* bias,
                  const float* post, float* C, int M, int K, int Nout, int act) {
    dim3 g((M + BM - 1) / BM, Nout / BM);
    gemm_wmma_kernel<<<g, 256, 0, stream>>>(A, nullptr, nullptr, nullptr, nullptr,
                                            nullptr, 0, W, bias, post, C, M, K, Nout, act);
  };
  auto gemm_cat = [&](const float* A0, const int* I0, const float* A1, const int* I1,
                      const float* A2, const int* I2, const float* W, const float* bias,
                      float* C, int M, int Nout, int act) {
    dim3 g((M + BM - 1) / BM, Nout / BM);
    gemm_wmma_kernel<<<g, 256, 0, stream>>>(A0, I0, A1, I1, A2, I2, 128,
                                            W, bias, nullptr, C, M, 384, Nout, act);
  };
  auto ln = [&](const float* in, const float* pre, const int* pidx, const float* g,
                const float* b, const float* post, float* out, int rows, int D, int act) {
    ln_act_kernel<<<rows, D, 0, stream>>>(in, pre, pidx, g, b, post, out, D, act);
  };

  // =========================== per-frame GNN ===============================
  for (int t = 0; t < T; t++) {
    float* h = nf + (size_t)t * N * 128;
    float* e = ef + (size_t)t * E * 128;
    const int* src = eidx + (size_t)t * 2 * E;
    const int* dst = src + E;
    const int* sid = snid + (size_t)t * N;

    gemm(x  + (size_t)t * N * 64, ne_w,   ne_b,   nullptr, h, N, 64, 128, 0);
    gemm(ea + (size_t)t * E * 32, eenc_w, eenc_b, nullptr, e, E, 32, 128, 0);

    for (int l = 0; l < 2; l++) {
      const int B = 12 + 22 * l;
      const float *a_b = F(B+0), *a_g = F(B+1), *a_ob = F(B+2), *a_ow = F(B+3);
      const float *a_we = F(B+4), *a_wk = F(B+5), *a_wq = F(B+6), *a_wv = F(B+7);
      const float *e_b1 = F(B+8), *e_b2 = F(B+9), *e_bb1 = F(B+10), *e_bb2 = F(B+11);
      const float *e_g1 = F(B+12), *e_g2 = F(B+13), *e_w1 = F(B+14), *e_w2 = F(B+15);
      const float *s_b = F(B+16), *s_b1 = F(B+17), *s_b2 = F(B+18);
      const float *s_g = F(B+19), *s_w1 = F(B+20), *s_w2 = F(B+21);

      // ---- edge attention ----
      gemm(h, a_wq, nullptr, nullptr, qb_, N, 128, 128, 0);
      gemm(h, a_wk, nullptr, nullptr, kb_, N, 128, 128, 0);
      gemm(h, a_wv, nullptr, nullptr, vb_, N, 128, 128, 0);
      gemm(e, a_we, nullptr, nullptr, eeb, E, 128, 128, 0);
      hipMemsetAsync(menc, 0, (size_t)N * 4 * sizeof(unsigned), stream);
      hipMemsetAsync(den,  0, (size_t)N * 4 * sizeof(float), stream);
      hipMemsetAsync(agg,  0, (size_t)N * 128 * sizeof(float), stream);
      edge_score_kernel<<<(E + 7) / 8, 256, 0, stream>>>(qb_, kb_, eeb, src, dst,
                                                         score, menc, E);
      edge_expsum_kernel<<<(E * 4 + 255) / 256, 256, 0, stream>>>(dst, score, menc, den, E);
      edge_aggregate_kernel<<<(E + 7) / 8, 256, 0, stream>>>(vb_, eeb, score, den,
                                                             src, dst, agg, E);
      gemm(agg, a_ow, a_ob, nullptr, tmpN, N, 128, 128, 0);
      ln(tmpN, h, nullptr, a_g, a_b, nullptr, h, N, 128, 1);   // gelu(LN(out + h))

      // ---- subnet pooling ----
      hipMemsetAsync(sf,  0, 64 * 128 * sizeof(float), stream);
      hipMemsetAsync(cnt, 0, 64 * sizeof(float), stream);
      subnet_accum_kernel<<<(N * 128 + 255) / 256, 256, 0, stream>>>(h, sid, sf, cnt, N);
      subnet_mean_kernel<<<(64 * 128) / 256, 256, 0, stream>>>(sf, cnt);
      gemm(sf,  s_w1, s_b1, nullptr, sft, 64, 128, 128, 2);    // relu
      gemm(sft, s_w2, s_b2, nullptr, sf2, 64, 128, 128, 0);
      ln(h, sf2, sid, s_g, s_b, nullptr, h, N, 128, 0);        // LN(h + sf2[sid])

      // ---- edge update (concat gather fused into GEMM) ----
      gemm_cat(h, src, h, dst, e, nullptr, e_w1, e_b1, eeb, E, 128, 0);
      ln(eeb, nullptr, nullptr, e_g1, e_bb1, nullptr, eeb, E, 128, 1);  // gelu(LN)
      gemm(eeb, e_w2, e_b2, nullptr, tmpE, E, 128, 128, 0);
      ln(tmpE, nullptr, nullptr, e_g2, e_bb2, e, e, E, 128, 1); // gelu(LN) + e
    }
  }

  // =========================== temporal stage ==============================
  conv_tpe_kernel<<<(N * 128 + 255) / 256, 256, 0, stream>>>(nf, tpe, conv_w, conv_b, xp, N);
  gemm(xp, ta_qw, ta_qb, nullptr, tq, T * N, 128, 128, 1);     // elu(.)+1
  gemm(xp, ta_kw, ta_kb, nullptr, tk, T * N, 128, 128, 1);     // elu(.)+1
  gemm(xp, ta_vw, ta_vb, nullptr, tv, T * N, 128, 128, 0);
  temporal_attn_kernel<<<(N * 4 + 7) / 8, 256, 0, stream>>>(tq, tk, tv, tat, N);
  float* nodeout = tq;                                          // reuse
  gemm(tat, ta_ow, ta_ob, xp, nodeout, T * N, 128, 128, 0);     // + xp residual

  // ============================ classifier =================================
  for (int t = 0; t < T; t++) {
    const float* e  = ef + (size_t)t * E * 128;
    const float* no = nodeout + (size_t)t * N * 128;
    const int* src = eidx + (size_t)t * 2 * E;
    const int* dst = src + E;
    gemm_cat(e, nullptr, no, src, no, dst, cls_w1, cls_b1, h1, E, 256, 0);
    ln(h1, nullptr, nullptr, cls_g, cls_bn, nullptr, h1, E, 256, 2);   // relu(LN)
    cls_out_kernel<<<(E + 7) / 8, 256, 0, stream>>>(
        h1, cls_w2, cls_b2, ((float*)d_out) + (size_t)t * E * 2, E);
  }
}
